// AttBlock_33062658244686
// MI455X (gfx1250) — matrix operand order repaired and checked by reading, not compiled, not run on hardware
//
#include <hip/hip_runtime.h>
#include <hip/hip_bf16.h>

// ---------------------------------------------------------------------------
// Problem constants (from reference)
// ---------------------------------------------------------------------------
#define NB       16      // batch
#define CIO      512     // in/out channels
#define LL       384     // sequence length
#define NH       8       // heads
#define DH       64      // head dim
#define CHID     512     // NH*DH
#define DG       8       // rezero group
#define ME       384     // MAXEMBED
#define KDIM     512     // reduction dim for both big GEMMs (CIO == CHID)

typedef __attribute__((ext_vector_type(16))) _Float16 v16h;
typedef __attribute__((ext_vector_type(8)))  float    v8f;

// ---------------------------------------------------------------------------
// CDNA5 async global->LDS copy (ASYNCcnt-tracked, no VGPR round trip).
// the LDS address is taken from the vdst VGPR; generic LDS pointers carry the LDS byte
// offset in their low 32 bits, so we can pass that directly.
// ---------------------------------------------------------------------------
__device__ __forceinline__ void async_g2l_b128(unsigned lds_off, const void* gptr) {
    asm volatile("global_load_async_to_lds_b128 %0, %1, off"
                 :: "v"(lds_off), "v"(gptr) : "memory");
}
__device__ __forceinline__ void wait_async0() {
    asm volatile("s_wait_asynccnt 0x0" ::: "memory");
}

// ---------------------------------------------------------------------------
// WMMA fragment helpers (wave32); this code's contiguous K order, same in A and B
//   A 16x32 f16 : lanes m / m+16 hold row M=m; K half split by lane-half.
//   element e -> K = (lane>=16 ? 16 : 0) + e
//   B 32x16 f16 : lane holds column N=lane&15; K = (lane>=16?16:0)+e.
// ---------------------------------------------------------------------------
__device__ __forceinline__ int a_koff(int e, int kh) {
    return kh * 16 + e;
}

// A(m,K) = p[K*ld + m]   (K-major A in memory, e.g. q^T for scores)
__device__ __forceinline__ v16h load_a_col(const _Float16* __restrict__ p, int ld, int lane) {
    int m = lane & 15, kh = (lane >> 4) & 1;
    v16h a;
#pragma unroll
    for (int e = 0; e < 16; ++e) a[e] = p[a_koff(e, kh) * ld + m];
    return a;
}

// B(K,n) = p[K*ld + n]   (K-major B, e.g. keys)
__device__ __forceinline__ v16h load_b_kmaj(const _Float16* __restrict__ p, int ld, int lane) {
    int nn = lane & 15, kh = (lane >> 4) & 1;
    v16h b;
#pragma unroll
    for (int e = 0; e < 16; ++e) b[e] = p[(kh * 16 + e) * ld + nn];
    return b;
}

// B(K,n) = p[n*ld + K]   (N-major B, e.g. V^T tiles: 16 contiguous f16/lane)
__device__ __forceinline__ v16h load_b_nmaj(const _Float16* __restrict__ p, int ld, int lane) {
    int nn = lane & 15, kh = (lane >> 4) & 1;
    v16h b;
#pragma unroll
    for (int e = 0; e < 16; ++e) b[e] = p[nn * ld + kh * 16 + e];
    return b;
}

#define WMMA_F16(A, B, C) \
    __builtin_amdgcn_wmma_f32_16x16x32_f16(false, (A), false, (B), (short)0, (C), false, false)

// ---------------------------------------------------------------------------
// Kernel 1: rezero fusion + f32->f16  (x0 = value path, x1 = qk path)
// ---------------------------------------------------------------------------
__global__ void prep_kernel(const float* __restrict__ x, const float* __restrict__ xorg,
                            const float* __restrict__ abspos,
                            const float* __restrict__ vres, const float* __restrict__ qkres,
                            const float* __restrict__ pres,
                            _Float16* __restrict__ x0h, _Float16* __restrict__ x1h, int total) {
    int idx = blockIdx.x * blockDim.x + threadIdx.x;
    if (idx >= total) return;
    int g = ((idx / LL) % CIO) / DG;
    float xv = x[idx], xo = xorg[idx], ap = abspos[idx];
    x0h[idx] = (_Float16)(xv + xo * vres[g]);
    x1h[idx] = (_Float16)(xv + xo * qkres[g] + ap * pres[g]);
}

// ---------------------------------------------------------------------------
// Kernel 2: f32 -> f16 weight conversion
// ---------------------------------------------------------------------------
__global__ void f2h_kernel(const float* __restrict__ in, _Float16* __restrict__ out, int n) {
    int idx = blockIdx.x * blockDim.x + threadIdx.x;
    if (idx < n) out[idx] = (_Float16)in[idx];
}

// ---------------------------------------------------------------------------
// Kernel 3: gate = gate_w @ x + gate_b   (tiny: 8x512 per column, f32)
// ---------------------------------------------------------------------------
__global__ void gate_kernel(const float* __restrict__ x, const float* __restrict__ gw,
                            const float* __restrict__ gb, float* __restrict__ gate) {
    int idx = blockIdx.x * blockDim.x + threadIdx.x;
    if (idx >= NB * NH * LL) return;
    int l = idx % LL, h = (idx / LL) % NH, n = idx / (LL * NH);
    const float* xp = x + (size_t)n * CIO * LL + l;
    const float* wp = gw + h * CIO;
    float acc = gb[h];
#pragma unroll 8
    for (int c = 0; c < CIO; ++c) acc += wp[c] * xp[c * LL];
    gate[idx] = acc;
}

// ---------------------------------------------------------------------------
// Kernel 4: batched GEMM  out[b] = W(Mdim x 512) * X[b](512 x LL)
// Block = 64x64 output tile. Per 32-wide K step, A(64x32) and B(32x64) f16
// panels are staged global->LDS with async B128 copies (16B/lane), then each
// wave runs two 16x16 WMMA tiles from LDS fragments.
// ---------------------------------------------------------------------------
template <bool F16OUT>
__global__ __launch_bounds__(256) void gemm_kernel(const _Float16* __restrict__ W,
                                                   const _Float16* __restrict__ X,
                                                   void* __restrict__ outp,
                                                   const float* __restrict__ bias, int Mdim) {
    __shared__ _Float16 ldsA[64 * 32];   // [row][K]
    __shared__ _Float16 ldsB[32 * 64];   // [K][col]

    const int t    = threadIdx.x;
    const int lane = t & 31;
    const int wave = t >> 5;

    const int nbt = LL / 64;                         // 6 col blocks
    const int tpb = (Mdim / 64) * nbt;               // blocks per batch (48)
    const int b   = blockIdx.x / tpb;
    const int rem = blockIdx.x % tpb;
    const int mt0 = (rem / nbt) * 64;
    const int nt0 = (rem % nbt) * 64;

    // wave -> (m-tile, two n-tiles)
    const int tm  = wave >> 1;           // 0..3
    const int tn0 = (wave & 1) * 2;      // 0 or 2

    // async-copy assignments: 16 bytes (8 halfs) per lane
    const int arow = t >> 2, acoff = (t & 3) * 8;    // A: 64 rows x 32 K
    const int brow = t >> 3, bcoff = (t & 7) * 8;    // B: 32 K x 64 cols
    const unsigned offA = (unsigned)(size_t)&ldsA[arow * 32 + acoff];
    const unsigned offB = (unsigned)(size_t)&ldsB[brow * 64 + bcoff];
    const _Float16* gA = W + (size_t)(mt0 + arow) * KDIM + acoff;
    const _Float16* gB = X + (size_t)b * KDIM * LL + (size_t)brow * LL + nt0 + bcoff;

    const int m = lane & 15, nn = lane & 15, kh = lane >> 4;
    v8f acc0 = {}, acc1 = {};

    for (int k0 = 0; k0 < KDIM; k0 += 32) {
        async_g2l_b128(offA, gA + k0);
        async_g2l_b128(offB, gB + (size_t)k0 * LL);
        wait_async0();
        __syncthreads();

        // A fragment: two contiguous 16B LDS reads per lane
        v16h a;
#pragma unroll
        for (int e = 0; e < 16; ++e) a[e] = ldsA[(tm * 16 + m) * 32 + a_koff(e, kh)];
        // B fragments for the wave's two n-tiles
        v16h b0, b1;
#pragma unroll
        for (int e = 0; e < 16; ++e) {
            int K = kh * 16 + e;
            b0[e] = ldsB[K * 64 + (tn0 + 0) * 16 + nn];
            b1[e] = ldsB[K * 64 + (tn0 + 1) * 16 + nn];
        }
        acc0 = WMMA_F16(a, b0, acc0);
        acc1 = WMMA_F16(a, b1, acc1);
        __syncthreads();   // protect LDS before next panel overwrite
    }

    const int half = lane >> 4;
    if constexpr (F16OUT) {
        _Float16* out = (_Float16*)outp + (size_t)b * Mdim * LL;
#pragma unroll
        for (int r = 0; r < 8; ++r) {
            int row = mt0 + tm * 16 + r + half * 8;
            out[row * LL + nt0 + (tn0 + 0) * 16 + nn] = (_Float16)acc0[r];
            out[row * LL + nt0 + (tn0 + 1) * 16 + nn] = (_Float16)acc1[r];
        }
    } else {
        float* out = (float*)outp + (size_t)b * Mdim * LL;
#pragma unroll
        for (int r = 0; r < 8; ++r) {
            int row = mt0 + tm * 16 + r + half * 8;
            out[row * LL + nt0 + (tn0 + 0) * 16 + nn] = acc0[r] + bias[row];
            out[row * LL + nt0 + (tn0 + 1) * 16 + nn] = acc1[r] + bias[row];
        }
    }
}

// ---------------------------------------------------------------------------
// Kernel 5: attention core. Block = (n, h, 128-row i-block); 8 waves, each
// owning a 16xL f32 score strip in LDS (8*16*384*4 = 192KB < 320KB WGP LDS).
// ---------------------------------------------------------------------------
__global__ __launch_bounds__(256) void attn_kernel(const _Float16* __restrict__ qh,
                                                   const _Float16* __restrict__ kh,
                                                   const _Float16* __restrict__ vh,
                                                   const float* __restrict__ gate,
                                                   const float* __restrict__ mask,
                                                   const float* __restrict__ norm,
                                                   const float* __restrict__ relpos,
                                                   _Float16* __restrict__ aoh) {
    extern __shared__ float smem[];
    const int lane = threadIdx.x & 31, wave = threadIdx.x >> 5;
    const int it = blockIdx.x % 3;
    const int h  = (blockIdx.x / 3) % NH;
    const int n  = blockIdx.x / (3 * NH);
    const int i0 = it * 128 + wave * 16;
    const int half = lane >> 4, nn = lane & 15;

    float* s_s  = smem + (size_t)wave * 16 * LL;
    float* s_m  = smem + 8 * 16 * LL + wave * 16;
    float* s_li = smem + 8 * 16 * LL + 8 * 16 + wave * 16;

    const _Float16* qb = qh + (size_t)(n * CHID + h * DH) * LL;
    const _Float16* kb = kh + (size_t)(n * CHID + h * DH) * LL;
    const _Float16* vb = vh + (size_t)(n * CHID + h * DH) * LL;

    const float invn = 1.0f / norm[n];
    const float* gaterow = gate + (size_t)(n * NH + h) * LL;
    const float* maskrow = mask + (size_t)n * LL;

    // ---- pass 1: scores ----
    for (int jt = 0; jt < LL / 16; ++jt) {
        const int j0 = jt * 16;
        __builtin_prefetch(kb + j0 + 16, 0, 1);          // global_prefetch next panel
        v8f c = {};
#pragma unroll
        for (int kk = 0; kk < DH; kk += 32) {
            v16h a  = load_a_col(qb + kk * LL + i0, LL, lane);
            v16h bf = load_b_kmaj(kb + kk * LL + j0, LL, lane);
            c = WMMA_F16(a, bf, c);
        }
        const int col = j0 + nn;
        const float bias_j = gaterow[col] + maskrow[col];
#pragma unroll
        for (int r = 0; r < 8; ++r) {
            int row  = i0 + r + half * 8;
            int ridx = ME - row + col;
            ridx = ridx < 0 ? 0 : (ridx > 2 * ME - 2 ? 2 * ME - 2 : ridx);
            s_s[(r + half * 8) * LL + col] = (c[r] + relpos[ridx] + bias_j) * invn;
        }
    }
    __syncthreads();

    // ---- pass 2: softmax stats (one row per lane, lanes 0..15) ----
    if (lane < 16) {
        const float* sr = s_s + lane * LL;
        float mx = -3.4e38f;
        for (int j = 0; j < LL; ++j) mx = fmaxf(mx, sr[j]);
        float l = 0.0f;
        for (int j = 0; j < LL; ++j) l += __expf(sr[j] - mx);
        s_m[lane]  = mx;
        s_li[lane] = 1.0f / l;
    }
    __syncthreads();

    // ---- pass 3: O = softmax(S) * v^T ----
    v8f o[4] = {{}, {}, {}, {}};
    const float rm = s_m[nn], rl = s_li[nn];
    for (int jt = 0; jt < LL / 32; ++jt) {
        const int j0 = jt * 32;
        v16h a;
#pragma unroll
        for (int e = 0; e < 16; ++e) {
            float s = s_s[nn * LL + j0 + a_koff(e, half)];
            a[e] = (_Float16)(__expf(s - rm) * rl);
        }
#pragma unroll
        for (int dt = 0; dt < 4; ++dt) {
            v16h bf = load_b_nmaj(vb + (dt * 16) * LL + j0, LL, lane);
            o[dt] = WMMA_F16(a, bf, o[dt]);
        }
    }
#pragma unroll
    for (int dt = 0; dt < 4; ++dt) {
#pragma unroll
        for (int r = 0; r < 8; ++r) {
            int d = h * DH + dt * 16 + nn;
            int i = i0 + r + half * 8;
            aoh[(size_t)(n * CHID + d) * LL + i] = (_Float16)o[dt][r];
        }
    }
}

// ---------------------------------------------------------------------------
// Host launcher
// ---------------------------------------------------------------------------
extern "C" void kernel_launch(void* const* d_in, const int* in_sizes, int n_in,
                              void* d_out, int out_size, void* d_ws, size_t ws_size,
                              hipStream_t stream) {
    const float* x       = (const float*)d_in[0];
    const float* xorg    = (const float*)d_in[1];
    const float* abspos  = (const float*)d_in[2];
    const float* mask    = (const float*)d_in[3];
    const float* norm    = (const float*)d_in[4];
    const float* qkpos   = (const float*)d_in[5];
    const float* qkorg   = (const float*)d_in[6];
    const float* vorg    = (const float*)d_in[7];
    const float* relpos  = (const float*)d_in[8];
    const float* gate_w  = (const float*)d_in[9];
    const float* gate_b  = (const float*)d_in[10];
    const float* q_w     = (const float*)d_in[11];
    const float* k_w     = (const float*)d_in[12];
    const float* v_w     = (const float*)d_in[13];
    const float* dense_w = (const float*)d_in[14];
    const float* dense_b = (const float*)d_in[15];
    float* out = (float*)d_out;

    const size_t ACT = (size_t)NB * CIO * LL;           // 3,145,728 elems
    const size_t WSZ = (size_t)CIO * CHID;              // 262,144 elems
    char* ws = (char*)d_ws;
    _Float16* x0h  = (_Float16*)(ws);
    _Float16* x1h  = (_Float16*)(ws + 2 * ACT);
    _Float16* qh   = (_Float16*)(ws + 4 * ACT);
    _Float16* khp  = (_Float16*)(ws + 6 * ACT);
    _Float16* vhp  = (_Float16*)(ws + 8 * ACT);
    _Float16* aoh  = (_Float16*)(ws + 10 * ACT);
    float*    gate = (float*)   (ws + 12 * ACT);
    char* wbase = ws + 12 * ACT + (size_t)NB * NH * LL * 4;
    _Float16* wqh = (_Float16*)(wbase);
    _Float16* wkh = (_Float16*)(wbase + 2 * WSZ);
    _Float16* wvh = (_Float16*)(wbase + 4 * WSZ);
    _Float16* wdh = (_Float16*)(wbase + 6 * WSZ);

    prep_kernel<<<(int)(ACT / 256), 256, 0, stream>>>(x, xorg, abspos, vorg, qkorg, qkpos,
                                                      x0h, x1h, (int)ACT);
    f2h_kernel<<<(int)(WSZ / 256), 256, 0, stream>>>(q_w,     wqh, (int)WSZ);
    f2h_kernel<<<(int)(WSZ / 256), 256, 0, stream>>>(k_w,     wkh, (int)WSZ);
    f2h_kernel<<<(int)(WSZ / 256), 256, 0, stream>>>(v_w,     wvh, (int)WSZ);
    f2h_kernel<<<(int)(WSZ / 256), 256, 0, stream>>>(dense_w, wdh, (int)WSZ);
    gate_kernel<<<(NB * NH * LL) / 256, 256, 0, stream>>>(x, gate_w, gate_b, gate);

    // 64x64-tile GEMM blocks: 16 * (512/64) * (384/64) = 768
    const int gemm_blocks = NB * (CHID / 64) * (LL / 64);
    gemm_kernel<true><<<gemm_blocks, 256, 0, stream>>>(wqh, x1h, (void*)qh,  nullptr, CHID);
    gemm_kernel<true><<<gemm_blocks, 256, 0, stream>>>(wkh, x1h, (void*)khp, nullptr, CHID);
    gemm_kernel<true><<<gemm_blocks, 256, 0, stream>>>(wvh, x0h, (void*)vhp, nullptr, CHID);

    const size_t shmem = (size_t)8 * 16 * LL * 4 + 2 * 8 * 16 * 4;
    attn_kernel<<<NB * NH * 3, 256, shmem, stream>>>(qh, khp, vhp, gate, mask, norm,
                                                     relpos, aoh);

    gemm_kernel<false><<<gemm_blocks, 256, 0, stream>>>(wdh, aoh, (void*)out, dense_b, CIO);
}
